// OSRAttention_13065290514877
// MI455X (gfx1250) — matrix operand order repaired and checked by reading, not compiled, not run on hardware
//
#include <hip/hip_runtime.h>
#include <math.h>

typedef __attribute__((ext_vector_type(16))) _Float16 v16h;
typedef __attribute__((ext_vector_type(8)))  float    v8f;

#define DIMC     256
#define HEADS    8
#define HEAD_DIM 32
#define IMH      64
#define IMW      64
#define NQ       4096
#define HKV      32
#define WKV      32
#define NKV      1024
#define BATCH    2
#define EPSF     1e-5f
#define SCALEF   0.17677669529663687f   /* 32^-0.5 */

// WMMA fragment layouts (ISA §7.12.2, wave32):
//  A (16x32 f16): lane l<16: row=l,   K = {0..7,16..23};  l>=16: row=l-16, K={8..15,24..31}
//    inverse: lane = m + 16*((k>>3)&1) ; j = (k&7) + 8*(k>=16)
//  B (32x16 f16): lane l<16: col=l,   K = {0..7,16..23};  l>=16: col=l-16, K={8..15,24..31}
//    inverse: lane = n + 16*((k>>3)&1) ; j = (k&7) + 8*(k>=16)
//  C (16x16 f32): vgpr r: row = r + 8*(lane>=16) ; col = lane&15

// ---------------------------------------------------------------------------
// Stage 1a: depthwise 5x5 stride-2 pad-2 + BN1 + ReLU + scale + BN2 -> t
// ---------------------------------------------------------------------------
__global__ void kv_sr_kernel(const float* __restrict__ x,
                             const float* __restrict__ dww,
                             const float* __restrict__ bn1g, const float* __restrict__ bn1b,
                             const float* __restrict__ bn1m, const float* __restrict__ bn1v,
                             const float* __restrict__ srpw,
                             const float* __restrict__ bn2g, const float* __restrict__ bn2b,
                             const float* __restrict__ bn2m, const float* __restrict__ bn2v,
                             float* __restrict__ t) {
    int idx = blockIdx.x * blockDim.x + threadIdx.x;
    if (idx >= BATCH * DIMC * HKV * WKV) return;
    int ox = idx & 31;
    int oy = (idx >> 5) & 31;
    int c  = (idx >> 10) & 255;
    int b  = idx >> 18;
    float acc = 0.f;
#pragma unroll
    for (int ky = 0; ky < 5; ++ky) {
        int iy = oy * 2 - 2 + ky;
        if (iy < 0 || iy >= IMH) continue;
#pragma unroll
        for (int kx = 0; kx < 5; ++kx) {
            int ix = ox * 2 - 2 + kx;
            if (ix < 0 || ix >= IMW) continue;
            acc += x[(((size_t)b * DIMC + c) * IMH + iy) * IMW + ix] * dww[c * 25 + ky * 5 + kx];
        }
    }
    float inv1 = bn1g[c] * rsqrtf(bn1v[c] + EPSF);
    float y = acc * inv1 + (bn1b[c] - bn1m[c] * inv1);
    y = fmaxf(y, 0.f) * srpw[c];
    float inv2 = bn2g[c] * rsqrtf(bn2v[c] + EPSF);
    y = y * inv2 + (bn2b[c] - bn2m[c] * inv2);
    t[idx] = y;
}

// ---------------------------------------------------------------------------
// Stage 1b: local depthwise 3x3 pad-1 + bias + residual -> kvfeat
// ---------------------------------------------------------------------------
__global__ void kv_local_kernel(const float* __restrict__ t,
                                const float* __restrict__ lw,
                                const float* __restrict__ lb,
                                float* __restrict__ kvfeat) {
    int idx = blockIdx.x * blockDim.x + threadIdx.x;
    if (idx >= BATCH * DIMC * HKV * WKV) return;
    int ox = idx & 31;
    int oy = (idx >> 5) & 31;
    int c  = (idx >> 10) & 255;
    int b  = idx >> 18;
    float acc = 0.f;
#pragma unroll
    for (int ky = 0; ky < 3; ++ky) {
        int iy = oy - 1 + ky;
        if (iy < 0 || iy >= HKV) continue;
#pragma unroll
        for (int kx = 0; kx < 3; ++kx) {
            int ix = ox - 1 + kx;
            if (ix < 0 || ix >= WKV) continue;
            acc += t[(((size_t)b * DIMC + c) * HKV + iy) * WKV + ix] * lw[c * 9 + ky * 3 + kx];
        }
    }
    kvfeat[idx] = acc + lb[c] + t[idx];
}

// ---------------------------------------------------------------------------
// Stage 2: q projection GEMM [256,256] x [256, B*4096] via WMMA f16.
// Output scattered directly into A-fragment layout qA[bh][qtile][lane][16].
// ---------------------------------------------------------------------------
__global__ __launch_bounds__(32) void qproj_kernel(const float* __restrict__ x,
                                                   const float* __restrict__ qw,
                                                   const float* __restrict__ qb,
                                                   _Float16* __restrict__ qA) {
    int tile = blockIdx.x;            // 16 m-tiles x 512 n-tiles
    int m0 = (tile & 15) * 16;        // output channel tile
    int n0 = (tile >> 4) * 16;        // spatial tile over B*4096
    int lane  = threadIdx.x & 31;
    int mA    = lane & 15;
    int koffA = (lane < 16) ? 0 : 8;
    int nB    = lane & 15;
    int koffB = (lane < 16) ? 0 : 16;
    int ng = n0 + nB;
    int b  = ng >> 12;
    int sp = ng & 4095;

    v8f acc = {};
    for (int kk = 0; kk < DIMC; kk += 32) {
        v16h A, Bm;
#pragma unroll
        for (int j = 0; j < 16; ++j) {
            int k = 2 * koffA + j;
            A[j] = (_Float16)qw[(size_t)(m0 + mA) * DIMC + kk + k];
        }
#pragma unroll
        for (int j = 0; j < 16; ++j) {
            int k = kk + koffB + j;
            Bm[j] = (_Float16)x[((size_t)b * DIMC + k) * NQ + sp];
        }
        acc = __builtin_amdgcn_wmma_f32_16x16x32_f16(false, A, false, Bm,
                                                     (short)0, acc, false, false);
    }
    int mhalf = (lane < 16) ? 0 : 8;
#pragma unroll
    for (int r = 0; r < 8; ++r) {
        int c = m0 + r + mhalf;                               // output channel
        float val = acc[r] + qb[c];
        int h = c >> 5, d = c & 31;
        int bh = b * HEADS + h;
        int qt = sp >> 4, m = sp & 15;                        // query tile / row
        int laneA = m + (((d >> 3) & 1) << 4);                // A-frag inverse map
        int jA = (d & 7) + ((d >= 16) ? 8 : 0);
        qA[((((size_t)bh * (NQ / 16) + qt) * 32 + laneA) << 4) + jA] = (_Float16)val;
    }
}

// ---------------------------------------------------------------------------
// Stage 3: kv projection GEMM [512,256] x [256, B*1024] via WMMA f16.
// k scattered into B-fragments kB[bh][ntile][lane][16]  (K-dim = head dim d)
// v scattered into B-fragments vB[bh][kchunk][nhalf][lane][16] (K-dim = kv pos)
// ---------------------------------------------------------------------------
__global__ __launch_bounds__(32) void kvproj_kernel(const float* __restrict__ kvfeat,
                                                    const float* __restrict__ kvw,
                                                    const float* __restrict__ kvb,
                                                    _Float16* __restrict__ kB,
                                                    _Float16* __restrict__ vB) {
    int tile = blockIdx.x;            // 32 m-tiles x 128 n-tiles
    int m0 = (tile & 31) * 16;        // output channel tile (0..511)
    int n0 = (tile >> 5) * 16;        // spatial tile over B*1024
    int lane  = threadIdx.x & 31;
    int mA    = lane & 15;
    int koffA = (lane < 16) ? 0 : 8;
    int nBl   = lane & 15;
    int koffB = (lane < 16) ? 0 : 16;
    int ng = n0 + nBl;
    int b  = ng >> 10;
    int sp = ng & 1023;

    v8f acc = {};
    for (int kk = 0; kk < DIMC; kk += 32) {
        v16h A, Bm;
#pragma unroll
        for (int j = 0; j < 16; ++j) {
            int k = 2 * koffA + j;
            A[j] = (_Float16)kvw[(size_t)(m0 + mA) * DIMC + kk + k];
        }
#pragma unroll
        for (int j = 0; j < 16; ++j) {
            int k = kk + koffB + j;
            Bm[j] = (_Float16)kvfeat[((size_t)b * DIMC + k) * NKV + sp];
        }
        acc = __builtin_amdgcn_wmma_f32_16x16x32_f16(false, A, false, Bm,
                                                     (short)0, acc, false, false);
    }
    int mhalf = (lane < 16) ? 0 : 8;
#pragma unroll
    for (int r = 0; r < 8; ++r) {
        int o = m0 + r + mhalf;                               // 0..511
        float val = acc[r] + kvb[o];
        if (o < DIMC) {                                       // ---- k ----
            int h = o >> 5, d = o & 31;
            int bh = b * HEADS + h;
            int nt = sp >> 4, n = sp & 15;                    // kv tile / col
            int laneB = n + (((d >> 3) & 1) << 4);            // B-frag inverse (K=d)
            int jB = (d & 7) + ((d >= 16) ? 8 : 0);
            kB[((((size_t)bh * (NKV / 16) + nt) * 32 + laneB) << 4) + jB] = (_Float16)val;
        } else {                                              // ---- v ----
            int c = o - DIMC;
            int h = c >> 5, d = c & 31;
            int bh = b * HEADS + h;
            int kc = sp >> 5, kk2 = sp & 31;                  // kv chunk / row-in-chunk
            int nh = d >> 4, dc = d & 15;                     // n-half / col
            int laneB = dc + (((kk2 >> 3) & 1) << 4);         // B-frag inverse (K=kv)
            int jB = (kk2 & 7) + ((kk2 >= 16) ? 8 : 0);
            vB[(((((size_t)bh * (NKV / 32) + kc) * 2 + nh) * 32 + laneB) << 4) + jB]
                = (_Float16)val;
        }
    }
}

// ---------------------------------------------------------------------------
// Stage 4: attention. One wave per (b,h, 16-query-row tile) -> 4096 blocks.
// Softmax max/sum computed ONLINE in registers during the QK^T phase (each
// lane owns 8 rows x 64 columns of S), then merged across the 16 lanes of
// each half-wave with a log-sum-exp shfl_xor reduction. No serial LDS
// re-read pass. All WMMA operands are single aligned 32B fragments.
// ---------------------------------------------------------------------------
__global__ __launch_bounds__(32) void attn_kernel(const v16h* __restrict__ qA,
                                                  const v16h* __restrict__ kB,
                                                  const v16h* __restrict__ vB,
                                                  const float* __restrict__ bias,
                                                  float* __restrict__ out) {
    __shared__ _Float16 S[16 * NKV];          // 32 KB, A-fragment layout
    __shared__ float rowmax[16];
    __shared__ float rowinv[16];

    int wg = blockIdx.x;
    int qt = wg & 255;
    int q0 = qt * 16;
    int bh = wg >> 8;                         // b*8 + h
    int h  = bh & 7;
    int b  = bh >> 3;
    int lane  = threadIdx.x & 31;
    int nB    = lane & 15;
    int mhalf = (lane < 16) ? 0 : 8;

    // Q tile: one fragment load
    v16h Aq = qA[((size_t)bh * (NQ / 16) + qt) * 32 + lane];

    // per-lane online softmax state for rows mhalf..mhalf+7 (column subset nB+16t)
    float mx[8], sm[8];
#pragma unroll
    for (int r = 0; r < 8; ++r) { mx[r] = -3.4e38f; sm[r] = 0.f; }

    // ---- S = (Q K^T)*scale + bias, scattered into A-layout LDS ----
    for (int nt = 0; nt < NKV / 16; ++nt) {
        int n0 = nt * 16;
        v16h Bk = kB[((size_t)bh * (NKV / 16) + nt) * 32 + lane];
        if (nt + 1 < NKV / 16)                // prefetch next bias chunk (HBM stream)
            __builtin_prefetch(&bias[((size_t)h * NQ + q0 + mhalf) * NKV + n0 + 16 + nB], 0, 1);
        v8f s = {};
        s = __builtin_amdgcn_wmma_f32_16x16x32_f16(false, Aq, false, Bk,
                                                   (short)0, s, false, false);
#pragma unroll
        for (int r = 0; r < 8; ++r) {
            int m = r + mhalf;
            float bv  = bias[((size_t)h * NQ + q0 + m) * NKV + n0 + nB];
            float val = s[r] * SCALEF + bv;
            // online log-sum-exp update
            float mo = mx[r];
            float mn = fmaxf(mo, val);
            sm[r] = sm[r] * __expf(mo - mn) + __expf(val - mn);
            mx[r] = mn;
            // scatter into A-fragment LDS slot
            int n  = n0 + nB;
            int kc = n >> 5, kk = n & 31;
            int laneS = m + (((kk >> 3) & 1) << 4);
            int jS = (kk & 7) + ((kk >= 16) ? 8 : 0);
            S[((kc * 32 + laneS) << 4) + jS] = (_Float16)val;
        }
    }

    // ---- merge partials across the 16 lanes of each half-wave ----
#pragma unroll
    for (int off = 1; off < 16; off <<= 1) {
#pragma unroll
        for (int r = 0; r < 8; ++r) {
            float omx = __shfl_xor(mx[r], off, 32);
            float osm = __shfl_xor(sm[r], off, 32);
            float M = fmaxf(mx[r], omx);
            sm[r] = sm[r] * __expf(mx[r] - M) + osm * __expf(omx - M);
            mx[r] = M;
        }
    }
    if (nB == 0) {                            // lane 0 -> rows 0..7, lane 16 -> rows 8..15
#pragma unroll
        for (int r = 0; r < 8; ++r) {
            rowmax[r + mhalf] = mx[r];
            rowinv[r + mhalf] = 1.f / sm[r];
        }
    }
    __syncthreads();

    // ---- in-place exp-normalize (A-layout: row = (addr>>4)&15) ----
    for (int i = lane; i < 16 * NKV; i += 32) {
        int m = (i >> 4) & 15;
        S[i] = (_Float16)(__expf((float)S[i] - rowmax[m]) * rowinv[m]);
    }
    __syncthreads();

    // ---- O = P V : fragment loads straight from LDS / global ----
    const v16h* Sv = (const v16h*)S;
    v8f o0 = {}, o1 = {};
    for (int kc = 0; kc < NKV / 32; ++kc) {
        v16h Ap  = Sv[kc * 32 + lane];
        v16h Bv0 = vB[(((size_t)bh * (NKV / 32) + kc) * 2 + 0) * 32 + lane];
        v16h Bv1 = vB[(((size_t)bh * (NKV / 32) + kc) * 2 + 1) * 32 + lane];
        o0 = __builtin_amdgcn_wmma_f32_16x16x32_f16(false, Ap, false, Bv0,
                                                    (short)0, o0, false, false);
        o1 = __builtin_amdgcn_wmma_f32_16x16x32_f16(false, Ap, false, Bv1,
                                                    (short)0, o1, false, false);
    }

    // ---- store: out[b][h*32+d][q] ----
#pragma unroll
    for (int r = 0; r < 8; ++r) {
        int q = q0 + r + mhalf;
        out[((size_t)b * DIMC + h * HEAD_DIM + nB) * NQ + q]      = o0[r];
        out[((size_t)b * DIMC + h * HEAD_DIM + 16 + nB) * NQ + q] = o1[r];
    }
}

// ---------------------------------------------------------------------------
extern "C" void kernel_launch(void* const* d_in, const int* in_sizes, int n_in,
                              void* d_out, int out_size, void* d_ws, size_t ws_size,
                              hipStream_t stream) {
    const float* x       = (const float*)d_in[0];
    const float* rpe     = (const float*)d_in[1];
    const float* q_w     = (const float*)d_in[2];
    const float* q_b     = (const float*)d_in[3];
    const float* kv_w    = (const float*)d_in[4];
    const float* kv_b    = (const float*)d_in[5];
    const float* sr_dw_w = (const float*)d_in[6];
    const float* bn1_g   = (const float*)d_in[7];
    const float* bn1_b   = (const float*)d_in[8];
    const float* bn1_m   = (const float*)d_in[9];
    const float* bn1_v   = (const float*)d_in[10];
    const float* sr_pw_w = (const float*)d_in[11];
    const float* bn2_g   = (const float*)d_in[12];
    const float* bn2_b   = (const float*)d_in[13];
    const float* bn2_m   = (const float*)d_in[14];
    const float* bn2_v   = (const float*)d_in[15];
    const float* local_w = (const float*)d_in[16];
    const float* local_b = (const float*)d_in[17];
    float* out = (float*)d_out;

    char* ws = (char*)d_ws;
    float*     t      = (float*)(ws);                        // 2 MB
    float*     kvfeat = (float*)(ws + (size_t)(2 << 20));    // 2 MB
    _Float16*  qA     = (_Float16*)(ws + (size_t)(4 << 20)); // 4 MB (A-fragments)
    _Float16*  kB     = (_Float16*)(ws + (size_t)(8 << 20)); // 1 MB (B-fragments)
    _Float16*  vB     = (_Float16*)(ws + (size_t)(9 << 20)); // 1 MB (B-fragments)

    const int elems = BATCH * DIMC * HKV * WKV;              // 524288
    kv_sr_kernel<<<(elems + 255) / 256, 256, 0, stream>>>(
        x, sr_dw_w, bn1_g, bn1_b, bn1_m, bn1_v, sr_pw_w,
        bn2_g, bn2_b, bn2_m, bn2_v, t);
    kv_local_kernel<<<(elems + 255) / 256, 256, 0, stream>>>(t, local_w, local_b, kvfeat);

    qproj_kernel<<<16 * 512, 32, 0, stream>>>(x, q_w, q_b, qA);
    kvproj_kernel<<<32 * 128, 32, 0, stream>>>(kvfeat, kv_w, kv_b, kB, vB);

    attn_kernel<<<BATCH * HEADS * (NQ / 16), 32, 0, stream>>>(
        (const v16h*)qA, (const v16h*)kB, (const v16h*)vB, rpe, out);
}
